// VQVAE_15006615733455
// MI455X (gfx1250) — matrix operand order repaired and checked by reading, not compiled, not run on hardware
//
#include <hip/hip_runtime.h>

// ---------------------------------------------------------------------------
// VQ-VAE forward for MI455X (gfx1250, wave32, WMMA).
// All GEMM-shaped layers use v_wmma_f32_16x16x32_bf16 (bf16 in, f32 accum).
// conv2 additionally stages its shared A tile in LDS via double-buffered
// global_load_async_to_lds_b128 (ASYNCcnt) -- one copy per block instead of
// one per wave. Activations staged in padded NHWC bf16 so implicit-GEMM
// fragment loads are contiguous b128s matching the CDNA5 WMMA VGPR layouts.
// Workspace arena (reused by lifetime), requires ws_size >= ~437 MB.
// ---------------------------------------------------------------------------

typedef unsigned short u16;
typedef __attribute__((ext_vector_type(8)))  __bf16 bhalf8;
typedef __attribute__((ext_vector_type(16))) __bf16 bhalf16;
typedef __attribute__((ext_vector_type(8)))  float  floatx8;

static __device__ __forceinline__ u16 f2bf(float f) {
  unsigned u = __builtin_bit_cast(unsigned, f);
  u += 0x7fffu + ((u >> 16) & 1u);          // round-to-nearest-even
  return (u16)(u >> 16);
}
static __device__ __forceinline__ float bf2f(u16 h) {
  unsigned u = ((unsigned)h) << 16;
  return __builtin_bit_cast(float, u);
}
static __device__ __forceinline__ bhalf8 ldb8(const u16* p) {
  return *reinterpret_cast<const bhalf8*>(p);
}
static __device__ __forceinline__ bhalf16 ldb16(const u16* p) {
  return *reinterpret_cast<const bhalf16*>(p);
}
static __device__ __forceinline__ bhalf16 cat8(bhalf8 a, bhalf8 b) {
  return __builtin_shufflevector(a, b, 0,1,2,3,4,5,6,7,8,9,10,11,12,13,14,15);
}
static __device__ __forceinline__ floatx8 wmma_bf16(bhalf16 a, bhalf16 b, floatx8 c) {
  // (neg_a, A, neg_b, B, c_mod, C, reuse_a, reuse_b)
  return __builtin_amdgcn_wmma_f32_16x16x32_bf16(false, a, false, b, (short)0, c,
                                                 false, false);
}
// async copy of 16 bytes global -> LDS, tracked by ASYNCcnt (gfx1250 VFLAT).
static __device__ __forceinline__ void async_cp16(const u16* g, u16* l) {
  unsigned lofs = (unsigned)(unsigned long long)(void*)l;   // LDS addr = addr[31:0]
  unsigned long long ga = (unsigned long long)(const void*)g;
  asm volatile("global_load_async_to_lds_b128 %0, %1, off"
               :: "v"(lofs), "v"(ga) : "memory");
}
static __device__ __forceinline__ void wait_async0() {
  asm volatile("s_wait_asynccnt 0" ::: "memory");
}

// ------------------------------- utilities ---------------------------------

__global__ void k_zero(uint4* __restrict__ p, long n4) {
  long g = (long)blockIdx.x * blockDim.x + threadIdx.x;
  if (g < n4) p[g] = make_uint4(0u, 0u, 0u, 0u);
}

// ------------------------------ weight prep --------------------------------

// w2 [128][128][4][4] -> w2g [o][tap*128+i] bf16 (tap-major K)
__global__ void k_prep_w2g(const float* __restrict__ w2, u16* __restrict__ w2g) {
  int g = blockIdx.x * 256 + threadIdx.x;
  if (g >= 128 * 2048) return;
  int o = g >> 11, k = g & 2047;
  int tap = k >> 7, i = k & 127;
  w2g[g] = f2bf(w2[(o * 128 + i) * 16 + tap]);
}
// w3 [64][128] -> bf16 (already [O][K])
__global__ void k_prep_w3b(const float* __restrict__ w3, u16* __restrict__ w3b) {
  int g = blockIdx.x * 256 + threadIdx.x;
  if (g < 64 * 128) w3b[g] = f2bf(w3[g]);
}
// dw1 [64][128] (in,out) -> dw1t [o=128][i=64] bf16
__global__ void k_prep_dw1t(const float* __restrict__ dw1, u16* __restrict__ dw1t) {
  int g = blockIdx.x * 256 + threadIdx.x;
  if (g >= 128 * 64) return;
  int o = g >> 6, i = g & 63;
  dw1t[g] = f2bf(dw1[i * 128 + o]);
}
// dw2 [128][64][4][4] -> per-parity-class wcb [cls][o=64][tap*128+i] bf16
__global__ void k_prep_wcb(const float* __restrict__ dw2, u16* __restrict__ wcb) {
  int g = blockIdx.x * 256 + threadIdx.x;
  if (g >= 4 * 64 * 512) return;
  int cls = g >> 15, rem = g & 32767;
  int o = rem >> 9, k = rem & 511;
  int tap = k >> 7, i = k & 127;
  int dy = tap >> 1, dx = tap & 1;
  int py = cls >> 1, px = cls & 1;
  int kh = (3 - py) - 2 * dy, kw = (3 - px) - 2 * dx;
  wcb[g] = f2bf(dw2[((i * 64 + o) * 4 + kh) * 4 + kw]);
}
__global__ void k_prep_embb(const float* __restrict__ emb, u16* __restrict__ embb) {
  int g = blockIdx.x * 256 + threadIdx.x;
  if (g < 512 * 64) embb[g] = f2bf(emb[g]);
}
__global__ void k_prep_ee(const float* __restrict__ emb, float* __restrict__ ee) {
  int j = blockIdx.x * 256 + threadIdx.x;
  if (j >= 512) return;
  float s = 0.f;
  #pragma unroll 8
  for (int d = 0; d < 64; ++d) { float v = emb[j * 64 + d]; s += v * v; }
  ee[j] = s;
}
// dw3 [64][3][3][3] (in,out,kH,kW) -> wd3g [o][tap=ty*3+tx][i] f32 (flipped)
__global__ void k_prep_wd3g(const float* __restrict__ dw3, float* __restrict__ wd3g) {
  int g = blockIdx.x * 256 + threadIdx.x;
  if (g >= 3 * 9 * 64) return;
  int o = g / 576, rem = g % 576;
  int tap = rem >> 6, i = rem & 63;
  int ty = tap / 3, tx = tap % 3;
  wd3g[g] = dw3[((i * 3 + o) * 3 + (2 - ty)) * 3 + (2 - tx)];
}
// x [32][3][256][256] -> padded xp [32][3][258][258] f32
__global__ void k_pad_x(const float* __restrict__ x, float* __restrict__ xp) {
  long g = (long)blockIdx.x * 256 + threadIdx.x;
  if (g >= (long)32 * 3 * 258 * 258) return;
  int wp = (int)(g % 258); long t = g / 258;
  int hp = (int)(t % 258); t /= 258;
  int c = (int)(t % 3); int n = (int)(t / 3);
  float v = 0.f;
  if (hp >= 1 && hp <= 256 && wp >= 1 && wp <= 256)
    v = x[(((long)n * 3 + c) * 256 + (hp - 1)) * 256 + (wp - 1)];
  xp[g] = v;
}

// ------------------------- encoder stage 1 (direct) ------------------------

// conv 3->128 k4 s2 p1 over padded xp; bias omitted (absorbed by following BN)
__global__ void k_conv1(const float* __restrict__ xp, const float* __restrict__ w1,
                        float* __restrict__ out) {
  __shared__ float ws[128 * 48];
  for (int i = threadIdx.x; i < 128 * 48; i += 256) ws[i] = w1[i];
  __syncthreads();
  int gid = blockIdx.x * 256 + threadIdx.x;       // 32*128*128 exact
  int ow = gid & 127, oh = (gid >> 7) & 127, n = gid >> 14;
  float p[48];
  #pragma unroll
  for (int c = 0; c < 3; ++c)
    #pragma unroll
    for (int kh = 0; kh < 4; ++kh)
      #pragma unroll
      for (int kw = 0; kw < 4; ++kw)
        p[c * 16 + kh * 4 + kw] =
            xp[(((long)n * 3 + c) * 258 + (2 * oh + kh)) * 258 + (2 * ow + kw)];
  for (int oc = 0; oc < 128; ++oc) {
    const float* wr = &ws[oc * 48];
    float a = 0.f;
    #pragma unroll
    for (int k = 0; k < 48; ++k) a += p[k] * wr[k];
    out[(((long)n * 128 + oc) * 128 + oh) * 128 + ow] = a;
  }
}

// -------------------------- BN stats / BN apply ----------------------------

// one block per channel; deterministic LDS tree reduction
__global__ void k_stats(const float* __restrict__ raw, float* __restrict__ sums,
                        float* __restrict__ sumsq, int C, int HW, int N) {
  int c = blockIdx.x, tid = threadIdx.x;
  float s = 0.f, q = 0.f;
  int per = N * HW;
  for (int j = tid; j < per; j += 256) {
    int n = j / HW, r = j - n * HW;
    float v = raw[((long)n * C + c) * HW + r];
    s += v; q += v * v;
  }
  __shared__ float ss[256], qq[256];
  ss[tid] = s; qq[tid] = q; __syncthreads();
  for (int st = 128; st > 0; st >>= 1) {
    if (tid < st) { ss[tid] += ss[tid + st]; qq[tid] += qq[tid + st]; }
    __syncthreads();
  }
  if (tid == 0) { sums[c] = ss[0]; sumsq[c] = qq[0]; }
}

// normalize (+ReLU) and repack NCHW f32 -> padded NHWC bf16
__global__ void k_bn_apply(const float* __restrict__ raw, const float* __restrict__ sums,
                           const float* __restrict__ sumsq, const float* __restrict__ gamma,
                           const float* __restrict__ beta, u16* __restrict__ out,
                           int C, int H, int Wd, int P, int N) {
  long gid = (long)blockIdx.x * 256 + threadIdx.x;
  long total = (long)N * C * H * Wd;
  if (gid >= total) return;
  int w = (int)(gid % Wd); long t = gid / Wd;
  int h = (int)(t % H); t /= H;
  int c = (int)(t % C); int n = (int)(t / C);
  float cnt = (float)N * H * Wd;
  float m  = sums[c] / cnt;
  float vr = sumsq[c] / cnt - m * m;
  float sc = gamma[c] * rsqrtf(vr + 1e-5f);
  float sh = beta[c] - m * sc;
  float v = fmaxf(raw[gid] * sc + sh, 0.f);
  int Hp = H + 2 * P, Wp = Wd + 2 * P;
  out[(((long)n * Hp + h + P) * Wp + (w + P)) * C + c] = f2bf(v);
}

// ---------------------- WMMA implicit-GEMM conv kernels --------------------

// conv2: 128->128 k4 s2 p1; A from padded NHWC h1p [32][130][130][128],
// B = w2g [128][2048]; out f32 NCHW [32][128][64][64].
// One M-tile per block; 8 waves = 8 oc tiles share one A tile per k-chunk,
// staged in LDS by double-buffered async global->LDS copies (ASYNCcnt).
__global__ void k_conv2(const u16* __restrict__ h1p, const u16* __restrict__ w2g,
                        float* __restrict__ out) {
  __shared__ __align__(16) u16 As[2][16 * 32];
  const int tid = threadIdx.x;
  const int lane = tid & 31, wv = tid >> 5;
  const int l16 = lane & 15, hi = lane >> 4;
  const int m0 = blockIdx.x * 16;                  // 8192 tiles
  const int n = m0 >> 12, oh = (m0 >> 6) & 63, w0 = m0 & 63;
  const u16* brow = w2g + (wv * 16 + l16) * 2048;
  const long nbase = (long)n * 130 * 130 * 128;

  // loader: threads 0..63 fetch the 16x32 bf16 A tile of one k-chunk (1 KB)
  const int lr = tid >> 2;              // row (ow offset) 0..15
  const int lc = (tid & 3) << 3;        // halfword offset 0,8,16,24
  const int low = w0 + lr;

  auto issue = [&](int kc, int buf) {
    if (tid < 64) {
      int tap = kc >> 2, kh = tap >> 2, kw = tap & 3;
      const u16* g = h1p + nbase +
                     (((long)(2 * oh + kh) * 130 + (2 * low + kw)) << 7) +
                     ((kc & 3) << 5) + lc;
      async_cp16(g, &As[buf][lr * 32 + lc]);
    }
  };

  issue(0, 0);
  wait_async0();
  __syncthreads();

  floatx8 acc = {};
  for (int kc = 0; kc < 64; ++kc) {
    const int cur = kc & 1;
    if (kc + 1 < 64) issue(kc + 1, cur ^ 1);
    const u16* ar = &As[cur][l16 * 32];
    bhalf16 a = cat8(*reinterpret_cast<const bhalf8*>(ar + (hi ? 8 : 0)),
                     *reinterpret_cast<const bhalf8*>(ar + 16 + (hi ? 8 : 0)));
    bhalf16 b = cat8(ldb8(brow + (kc << 5) + (hi ? 8 : 0)), ldb8(brow + (kc << 5) + 16 + (hi ? 8 : 0)));
    __builtin_prefetch(brow + ((kc + 1) << 5), 0, 1);
    acc = wmma_bf16(a, b, acc);
    wait_async0();        // next buffer's async copies complete
    __syncthreads();      // + all waves done reading before it is overwritten
  }
  const int oc = wv * 16 + l16;
  float* ob = out + (((long)n * 128 + oc) * 64 + oh) * 64 + w0 + (hi ? 8 : 0);
  #pragma unroll
  for (int r = 0; r < 8; ++r) ob[r] = acc[r];
}

// conv3: 1x1 128->64 (+bias); A = h2 NHWC [m][128]; out z f32 NCHW + zb bf16 NCHW
__global__ void k_conv3(const u16* __restrict__ h2, const u16* __restrict__ w3b,
                        const float* __restrict__ b3, float* __restrict__ z,
                        u16* __restrict__ zb) {
  const int lane = threadIdx.x & 31, wv = threadIdx.x >> 5;
  const int l16 = lane & 15, hi = lane >> 4;
  const int job = blockIdx.x * 8 + wv;             // 32768 wave-jobs
  const int mt = job >> 2, nt = job & 3;
  const int m0 = mt * 16;
  const int n = m0 >> 12, oh = (m0 >> 6) & 63, w0 = m0 & 63;
  const u16* arow = h2 + (long)(m0 + l16) * 128;
  const u16* brow = w3b + (nt * 16 + l16) * 128;
  floatx8 acc = {};
  #pragma unroll
  for (int kc = 0; kc < 4; ++kc) {
    bhalf16 a = cat8(ldb8(arow + kc * 32 + hi * 8), ldb8(arow + kc * 32 + 16 + hi * 8));
    bhalf16 b = cat8(ldb8(brow + kc * 32 + hi * 8), ldb8(brow + kc * 32 + 16 + hi * 8));
    acc = wmma_bf16(a, b, acc);
  }
  const int oc = nt * 16 + l16;
  const float bias = b3[oc];
  long ob = (((long)n * 64 + oc) * 64 + oh) * 64 + w0 + (hi ? 8 : 0);
  #pragma unroll
  for (int r = 0; r < 8; ++r) {
    float v = acc[r] + bias;
    z[ob + r] = v;
    zb[ob + r] = f2bf(v);
  }
}

// -------------------------------- VQ stage ---------------------------------

// rows m2=(n,c,h) of the raw NCHW view (64 W pixels each); GEMM vs 512 codes,
// cross-lane argmin, codebook gather -> zq NHWC bf16, idx + loss partials.
__global__ void k_vq(const u16* __restrict__ zb, const float* __restrict__ z,
                     const u16* __restrict__ embb, const float* __restrict__ ee,
                     const float* __restrict__ emb, u16* __restrict__ zq,
                     int* __restrict__ idx_out, float* __restrict__ partials) {
  const int lane = threadIdx.x & 31;
  const int l16 = lane & 15, hi = lane >> 4;
  const int wave = (blockIdx.x * 256 + threadIdx.x) >> 5;  // 0..8191
  const int m0 = wave * 16;
  const u16* arow = zb + (long)(m0 + l16) * 64;
  bhalf16 a0 = cat8(ldb8(arow + hi * 8), ldb8(arow + 16 + hi * 8));
  bhalf16 a1 = cat8(ldb8(arow + 32 + hi * 8), ldb8(arow + 48 + hi * 8));
  float bestv[8]; int besti[8];
  #pragma unroll
  for (int r = 0; r < 8; ++r) { bestv[r] = 3.4e38f; besti[r] = 0; }
  for (int jt = 0; jt < 32; ++jt) {
    int j = jt * 16 + l16;
    const u16* brow = embb + j * 64;
    bhalf16 b0 = cat8(ldb8(brow + hi * 8), ldb8(brow + 16 + hi * 8));
    bhalf16 b1 = cat8(ldb8(brow + 32 + hi * 8), ldb8(brow + 48 + hi * 8));
    floatx8 acc = {};
    acc = wmma_bf16(a0, b0, acc);
    acc = wmma_bf16(a1, b1, acc);
    float ej = ee[j];
    #pragma unroll
    for (int r = 0; r < 8; ++r) {
      float dv = ej - 2.f * acc[r];                // zz[m] constant per row
      if (dv < bestv[r] || (dv == bestv[r] && j < besti[r])) {
        bestv[r] = dv; besti[r] = j;
      }
    }
  }
  // reduce across the 16 lanes of each half (rows r / 8+r)
  #pragma unroll
  for (int r = 0; r < 8; ++r) {
    float v = bestv[r]; int bi = besti[r];
    #pragma unroll
    for (int msk = 8; msk >= 1; msk >>= 1) {
      float ov = __shfl_xor(v, msk, 32);
      int   oi = __shfl_xor(bi, msk, 32);
      if (ov < v || (ov == v && oi < bi)) { v = ov; bi = oi; }
    }
    bestv[r] = v; besti[r] = bi;
  }
  float lacc = 0.f;
  for (int mr = 0; mr < 16; ++mr) {
    int bi = __shfl(besti[mr & 7], (mr < 8) ? 0 : 16, 32);
    int m2 = m0 + mr;
    if (lane == 0) idx_out[m2] = bi;
    int n = m2 >> 12, c = (m2 >> 6) & 63, h = m2 & 63;
    const float* erow = emb + bi * 64;
    const float* zrow = z + (long)m2 * 64;
    #pragma unroll
    for (int t = 0; t < 2; ++t) {
      int d = lane * 2 + t;
      float q = erow[d];
      float dz = q - zrow[d];
      lacc += dz * dz;
      zq[(((long)n * 64 + h) * 64 + d) * 64 + c] = f2bf(q);
    }
  }
  #pragma unroll
  for (int msk = 16; msk >= 1; msk >>= 1) lacc += __shfl_xor(lacc, msk, 32);
  if (lane == 0) partials[wave] = lacc;
}

__global__ void k_reduce_loss(const float* __restrict__ partials, float* __restrict__ out) {
  __shared__ float sh[256];
  float s = 0.f;
  for (int j = threadIdx.x; j < 8192; j += 256) s += partials[j];
  sh[threadIdx.x] = s; __syncthreads();
  for (int st = 128; st > 0; st >>= 1) {
    if (threadIdx.x < st) sh[threadIdx.x] += sh[threadIdx.x + st];
    __syncthreads();
  }
  // loss = MSE + 0.25*MSE (both terms identical in forward value)
  if (threadIdx.x == 0) out[0] = 1.25f * sh[0] / 8388608.f;
}

// ------------------------------- decoder -----------------------------------

// d1: 1x1 convT 64->128 (+bias); A = zq NHWC [m][64]; out padded NHWC bf16 h3p
__global__ void k_d1(const u16* __restrict__ zq, const u16* __restrict__ dw1t,
                     const float* __restrict__ db1, u16* __restrict__ h3p) {
  const int lane = threadIdx.x & 31, wv = threadIdx.x >> 5;
  const int l16 = lane & 15, hi = lane >> 4;
  const int m0 = blockIdx.x * 16;                  // 8192 tiles, wave = oc tile
  const int n = m0 >> 12, h = (m0 >> 6) & 63, w0 = m0 & 63;
  const u16* arow = zq + (long)(m0 + l16) * 64;
  const u16* brow = dw1t + (wv * 16 + l16) * 64;
  floatx8 acc = {};
  #pragma unroll
  for (int kc = 0; kc < 2; ++kc) {
    bhalf16 a = cat8(ldb8(arow + kc * 32 + hi * 8), ldb8(arow + kc * 32 + 16 + hi * 8));
    bhalf16 b = cat8(ldb8(brow + kc * 32 + hi * 8), ldb8(brow + kc * 32 + 16 + hi * 8));
    acc = wmma_bf16(a, b, acc);
  }
  const int oc = wv * 16 + l16;
  const float bias = db1[oc];
  #pragma unroll
  for (int r = 0; r < 8; ++r) {
    int w = w0 + (hi ? 8 : 0) + r;
    h3p[(((long)n * 66 + h + 1) * 66 + (w + 1)) * 128 + oc] = f2bf(acc[r] + bias);
  }
}

// d2: convT 128->64 k4 s2 p1 as 4 parity-class 2x2 stride-1 GEMMs over h3p;
// bias omitted (absorbed by following BN). out f32 NCHW [32][64][128][128].
__global__ void k_d2(const u16* __restrict__ h3p, const u16* __restrict__ wcb,
                     float* __restrict__ out) {
  const int lane = threadIdx.x & 31, wv = threadIdx.x >> 5;
  const int l16 = lane & 15, hi = lane >> 4;
  const int job = blockIdx.y * 8 + wv;             // 16 jobs: cls*4 + octile
  const int cls = job >> 2, oc0 = (job & 3) * 16;
  const int py = cls >> 1, px = cls & 1;
  const int m0 = blockIdx.x * 16;
  const int n = m0 >> 12, t = (m0 >> 6) & 63, s0 = m0 & 63;
  const int s = s0 + l16;
  const u16* brow = wcb + ((long)cls * 64 + oc0 + l16) * 512;
  const long nbase = (long)n * 66 * 66 * 128;
  floatx8 acc = {};
  #pragma unroll 4
  for (int kc = 0; kc < 16; ++kc) {
    int tap = kc >> 2;
    int dy = tap >> 1, dx = tap & 1;
    int ioff = ((kc & 3) << 5) + (hi ? 8 : 0);
    const u16* ap = h3p + nbase +
                    (((long)(t + py + dy) * 66 + (s + px + dx)) << 7) + ioff;
    __builtin_prefetch(brow + ((kc + 1) << 5), 0, 1);
    bhalf16 a = cat8(ldb8(ap), ldb8(ap + 16));
    bhalf16 b = cat8(ldb8(brow + (kc << 5) + (hi ? 8 : 0)), ldb8(brow + (kc << 5) + 16 + (hi ? 8 : 0)));
    acc = wmma_bf16(a, b, acc);
  }
  const int oc = oc0 + l16;
  const int oh = 2 * t + py;
  #pragma unroll
  for (int r = 0; r < 8; ++r) {
    int ow = 2 * (s0 + (hi ? 8 : 0) + r) + px;
    out[(((long)n * 64 + oc) * 128 + oh) * 128 + ow] = acc[r];
  }
}

// d3: convT 64->3 k3 s1 p1 (direct, tiny N) + tanh -> recon
__global__ void k_d3(const u16* __restrict__ h4p, const float* __restrict__ wd3g,
                     const float* __restrict__ db3, float* __restrict__ recon) {
  __shared__ float ws[3 * 9 * 64];
  __shared__ float bs[3];
  for (int i = threadIdx.x; i < 1728; i += 256) ws[i] = wd3g[i];
  if (threadIdx.x < 3) bs[threadIdx.x] = db3[threadIdx.x];
  __syncthreads();
  int gid = blockIdx.x * 256 + threadIdx.x;        // 32*128*128 exact
  int ow = gid & 127, oh = (gid >> 7) & 127, n = gid >> 14;
  float a0 = bs[0], a1 = bs[1], a2 = bs[2];
  for (int ty = 0; ty < 3; ++ty)
    for (int tx = 0; tx < 3; ++tx) {
      const u16* ip = h4p + (((long)n * 130 + oh + ty) * 130 + (ow + tx)) * 64;
      const int tb = (ty * 3 + tx) * 64;
      #pragma unroll 8
      for (int i = 0; i < 64; ++i) {
        float v = bf2f(ip[i]);
        a0 += v * ws[tb + i];
        a1 += v * ws[576 + tb + i];
        a2 += v * ws[1152 + tb + i];
      }
    }
  long base = ((long)n * 3) * 16384 + (long)oh * 128 + ow;
  recon[base]             = tanhf(a0);
  recon[base + 16384]     = tanhf(a1);
  recon[base + 2 * 16384] = tanhf(a2);
}

// ------------------------------- launcher ----------------------------------

extern "C" void kernel_launch(void* const* d_in, const int* in_sizes, int n_in,
                              void* d_out, int out_size, void* d_ws, size_t ws_size,
                              hipStream_t stream) {
  (void)in_sizes; (void)n_in; (void)out_size; (void)ws_size;
  const float* x     = (const float*)d_in[0];
  const float* emb   = (const float*)d_in[1];
  const float* w1    = (const float*)d_in[2];
  const float* g1    = (const float*)d_in[4];
  const float* be1   = (const float*)d_in[5];
  const float* w2    = (const float*)d_in[6];
  const float* g2    = (const float*)d_in[8];
  const float* be2   = (const float*)d_in[9];
  const float* w3    = (const float*)d_in[10];
  const float* b3    = (const float*)d_in[11];
  const float* dw1   = (const float*)d_in[12];
  const float* db1   = (const float*)d_in[13];
  const float* dw2   = (const float*)d_in[14];
  const float* dg    = (const float*)d_in[16];
  const float* dbeta = (const float*)d_in[17];
  const float* dw3   = (const float*)d_in[18];
  const float* db3   = (const float*)d_in[19];

  char* W = (char*)d_ws;
  const long MBL = 1024 * 1024;
  float* stats    = (float*)W;                       // 6x128 floats
  float* s1 = stats,       *q1 = stats + 128;
  float* s2 = stats + 256, *q2 = stats + 384;
  float* s3 = stats + 512, *q3 = stats + 576;
  float* partials = (float*)(W + (16 << 10));        // 8192 f32
  u16*  w2g  = (u16*)(W + (64 << 10));               // 512 KB
  u16*  w3b  = (u16*)(W + (64 << 10) + 0x80000);
  u16*  dw1t = (u16*)(W + (64 << 10) + 0x84000);
  u16*  wcb  = (u16*)(W + (64 << 10) + 0x88000);     // 256 KB
  u16*  embb = (u16*)(W + (64 << 10) + 0xC8000);
  float* ee  = (float*)(W + (64 << 10) + 0xD8000);
  float* wd3g= (float*)(W + (64 << 10) + 0xD9000);
  float* xp    = (float*)(W + 1 * MBL);              // 25.6 MB
  float* c1raw = (float*)(W + 32 * MBL);             // 256 MiB (reused later)
  u16*  h1p    = (u16*)(W + 304 * MBL);              // 132 MiB (reused by d2raw)
  float* c2raw = (float*)(W + 32 * MBL);             // 64 MiB
  u16*  h2     = (u16*)(W + 96 * MBL);               // 32 MiB
  float* z     = (float*)(W + 128 * MBL);            // 32 MiB
  u16*  zb     = (u16*)(W + 160 * MBL);              // 16 MiB
  u16*  zq     = (u16*)(W + 176 * MBL);              // 16 MiB
  u16*  h3p    = (u16*)(W + 192 * MBL);              // 34 MiB
  float* d2raw = (float*)(W + 304 * MBL);            // 128 MiB
  u16*  h4p    = (u16*)(W + 32 * MBL);               // 66 MiB

  float* recon    = (float*)d_out;
  float* loss_out = recon + 1572864;
  int*   idx_out  = (int*)(recon + 1572865);

  dim3 B(256);
  // weight prep
  k_prep_w2g <<<1024, B, 0, stream>>>(w2, w2g);
  k_prep_w3b <<<32,   B, 0, stream>>>(w3, w3b);
  k_prep_dw1t<<<32,   B, 0, stream>>>(dw1, dw1t);
  k_prep_wcb <<<512,  B, 0, stream>>>(dw2, wcb);
  k_prep_embb<<<128,  B, 0, stream>>>(emb, embb);
  k_prep_ee  <<<2,    B, 0, stream>>>(emb, ee);
  k_prep_wd3g<<<7,    B, 0, stream>>>(dw3, wd3g);
  k_pad_x    <<<24962, B, 0, stream>>>(x, xp);
  // encoder
  k_conv1<<<2048, B, 0, stream>>>(xp, w1, c1raw);
  k_stats<<<128, B, 0, stream>>>(c1raw, s1, q1, 128, 16384, 32);
  k_zero <<<33800, B, 0, stream>>>((uint4*)h1p, 8652800L);
  k_bn_apply<<<262144, B, 0, stream>>>(c1raw, s1, q1, g1, be1, h1p, 128, 128, 128, 1, 32);
  k_conv2<<<8192, B, 0, stream>>>(h1p, w2g, c2raw);
  k_stats<<<128, B, 0, stream>>>(c2raw, s2, q2, 128, 4096, 32);
  k_bn_apply<<<65536, B, 0, stream>>>(c2raw, s2, q2, g2, be2, h2, 128, 64, 64, 0, 32);
  k_conv3<<<4096, B, 0, stream>>>(h2, w3b, b3, z, zb);
  // vector quantization
  k_vq<<<1024, B, 0, stream>>>(zb, z, embb, ee, emb, zq, idx_out, partials);
  k_reduce_loss<<<1, B, 0, stream>>>(partials, loss_out);
  // decoder
  k_zero<<<8713, B, 0, stream>>>((uint4*)h3p, 2230272L);
  k_d1<<<8192, B, 0, stream>>>(zq, dw1t, db1, h3p);
  k_d2<<<dim3(8192, 2), B, 0, stream>>>(h3p, wcb, d2raw);
  k_stats<<<64, B, 0, stream>>>(d2raw, s3, q3, 64, 16384, 32);
  k_zero<<<16901, B, 0, stream>>>((uint4*)h4p, 4326400L);
  k_bn_apply<<<131072, B, 0, stream>>>(d2raw, s3, q3, dg, dbeta, h4p, 64, 128, 128, 1, 32);
  k_d3<<<2048, B, 0, stream>>>(h4p, wd3g, db3, recon);
}
